// RITS_10419590660319
// MI455X (gfx1250) — matrix operand order repaired and checked by reading, not compiled, not run on hardware
//
#include <hip/hip_runtime.h>
#include <hip/hip_bf16.h>

typedef __attribute__((ext_vector_type(16))) __bf16 v16bf;
typedef __attribute__((ext_vector_type(8)))  float  v8f;

#define T_STEPS 100
#define FD 128
#define HD 256
#define NB 2048

// workspace byte offsets
#define ACC_OFF    0          // 400 f32 loss accumulators (num1,num2,num3,den per t)
#define WDX_OFF    2048       // 128 f32 diag(Wdx)
#define BSUM_OFF   2560       // 1024 f32 bih+bhh
#define WDH_OFF    8192       // [256][128] bf16
#define WHIST_OFF  73728      // [128][256] bf16
#define WFEAT_OFF  139264     // [128][128] bf16 (zero diag)
#define WCOMB_OFF  172032     // [128][256] bf16
#define WIH_OFF    237568     // [1024][256] bf16
#define WHH_OFF    761856     // [1024][256] bf16

__device__ inline v8f wmma_bf16(v16bf a, v16bf b, v8f c) {
  return __builtin_amdgcn_wmma_f32_16x16x32_bf16(false, a, false, b, (short)0, c,
                                                 false, false);
}

// A fragment (16x32 bf16) from row-major f32 LDS buffer, columns k0..k0+31.
// ISA layout: lane<16 holds row=lane, K {k0..k0+7, k0+16..k0+23};
//             lane>=16 holds row=lane-16, K {k0+8..k0+15, k0+24..k0+31}.
__device__ inline v16bf a_frag_f32(const float* buf, int stride, int k0, int lane) {
  int r  = lane & 15;
  int kb = k0 + ((lane >> 4) << 3);
  const float* p = buf + r * stride + kb;
  v16bf a;
#pragma unroll
  for (int i = 0; i < 8; ++i) a[i] = (__bf16)p[i];
#pragma unroll
  for (int i = 0; i < 8; ++i) a[8 + i] = (__bf16)p[16 + i];
  return a;
}

__device__ inline v16bf a_frag_bf(const __bf16* buf, int stride, int k0, int lane) {
  int r  = lane & 15;
  int kb = k0 + ((lane >> 4) << 3);
  const __bf16* p = buf + r * stride + kb;
  v16bf a;
#pragma unroll
  for (int i = 0; i < 8; ++i) a[i] = p[i];
#pragma unroll
  for (int i = 0; i < 8; ++i) a[8 + i] = p[16 + i];
  return a;
}

// A fragment for on-the-fly complement blend: m*x + (1-m)*e
__device__ inline v16bf a_frag_mix(const float* Mb, const float* Xb, const float* Eb,
                                   int stride, int k0, int lane) {
  int r    = lane & 15;
  int kb   = k0 + ((lane >> 4) << 3);
  int base = r * stride + kb;
  v16bf a;
#pragma unroll
  for (int i = 0; i < 8; ++i) {
    float mv = Mb[base + i];
    a[i] = (__bf16)(mv * Xb[base + i] + (1.f - mv) * Eb[base + i]);
  }
#pragma unroll
  for (int i = 0; i < 8; ++i) {
    float mv = Mb[base + 16 + i];
    a[8 + i] = (__bf16)(mv * Xb[base + 16 + i] + (1.f - mv) * Eb[base + 16 + i]);
  }
  return a;
}

// B fragment (32x16 bf16, K x N) from row-major bf16 weight W[N][K].
// ISA layout: lane<16 holds col=lane, K {k0..k0+7, k0+16..k0+23};
//             lane>=16 holds col=lane-16, K {k0+8..k0+15, k0+24..k0+31}.
__device__ inline v16bf b_frag(const __bf16* W, int K, int n0, int k0, int lane) {
  int n  = n0 + (lane & 15);
  int kb = k0 + ((lane >> 4) << 3);
  const __bf16* p = W + (size_t)n * K + kb;
  v16bf b;
#pragma unroll
  for (int i = 0; i < 8; ++i) b[i] = p[i];
#pragma unroll
  for (int i = 0; i < 8; ++i) b[8 + i] = p[16 + i];
  return b;
}

__device__ inline float sigm(float x) { return 1.0f / (1.0f + __expf(-x)); }

// ---------------------------------------------------------------------------
__global__ void rits_prep(const float* __restrict__ Wdh, const float* __restrict__ Whist,
                          const float* __restrict__ Wfeat, const float* __restrict__ Wcomb,
                          const float* __restrict__ Wih, const float* __restrict__ Whh,
                          const float* __restrict__ Wdx, const float* __restrict__ bih,
                          const float* __restrict__ bhh, char* __restrict__ ws) {
  float*  acc   = (float*)(ws + ACC_OFF);
  float*  wdx   = (float*)(ws + WDX_OFF);
  float*  bsum  = (float*)(ws + BSUM_OFF);
  __bf16* wdh   = (__bf16*)(ws + WDH_OFF);
  __bf16* whist = (__bf16*)(ws + WHIST_OFF);
  __bf16* wfeat = (__bf16*)(ws + WFEAT_OFF);
  __bf16* wcomb = (__bf16*)(ws + WCOMB_OFF);
  __bf16* wih   = (__bf16*)(ws + WIH_OFF);
  __bf16* whh   = (__bf16*)(ws + WHH_OFF);

  int tid = blockIdx.x * blockDim.x + threadIdx.x;
  int nt  = gridDim.x * blockDim.x;
  for (int i = tid; i < 400; i += nt) acc[i] = 0.f;
  for (int i = tid; i < FD; i += nt) wdx[i] = Wdx[i * FD + i];
  for (int i = tid; i < 4 * HD; i += nt) bsum[i] = bih[i] + bhh[i];
  for (int i = tid; i < HD * FD; i += nt) wdh[i] = (__bf16)Wdh[i];
  for (int i = tid; i < FD * HD; i += nt) whist[i] = (__bf16)Whist[i];
  for (int i = tid; i < FD * FD; i += nt) {
    int r = i >> 7, c = i & 127;
    wfeat[i] = (__bf16)((r == c) ? 0.f : Wfeat[i]);
  }
  for (int i = tid; i < FD * 2 * FD; i += nt) wcomb[i] = (__bf16)Wcomb[i];
  for (int i = tid; i < 4 * HD * 2 * FD; i += nt) wih[i] = (__bf16)Wih[i];
  for (int i = tid; i < 4 * HD * HD; i += nt) whh[i] = (__bf16)Whh[i];
}

// ---------------------------------------------------------------------------
__global__ __launch_bounds__(256)
void rits_main(const float* __restrict__ X, const float* __restrict__ Mm,
               const float* __restrict__ Dd, const float* __restrict__ bdh,
               const float* __restrict__ bdx, const float* __restrict__ bhist,
               const float* __restrict__ bfeat, const float* __restrict__ bcomb,
               char* __restrict__ ws, float* __restrict__ out) {
  __shared__ float  sX[16][FD];
  __shared__ float  sM[16][FD];
  __shared__ float  sGX[16][FD];
  __shared__ float  sXH[16][FD];   // holds x_h, then c_h (pointwise in-place)
  __shared__ float  sZH[16][FD];
  __shared__ float  sH[16][HD];
  __shared__ __bf16 sD[16][FD];

  const float*  wdx     = (const float*)(ws + WDX_OFF);
  const float*  bsum    = (const float*)(ws + BSUM_OFF);
  const __bf16* Wdh_b   = (const __bf16*)(ws + WDH_OFF);
  const __bf16* Whist_b = (const __bf16*)(ws + WHIST_OFF);
  const __bf16* Wfeat_b = (const __bf16*)(ws + WFEAT_OFF);
  const __bf16* Wcomb_b = (const __bf16*)(ws + WCOMB_OFF);
  const __bf16* Wih_b   = (const __bf16*)(ws + WIH_OFF);
  const __bf16* Whh_b   = (const __bf16*)(ws + WHH_OFF);
  float* acc_g = (float*)(ws + ACC_OFF);
  float* out_h = out + (size_t)NB * T_STEPS * FD;

  const int tid  = threadIdx.x;
  const int wave = tid >> 5, lane = tid & 31;
  const int b0   = blockIdx.x * 16;
  const int nlo  = lane & 15;
  const int mb   = (lane >> 4) << 3;

  for (int i = tid; i < 16 * HD; i += 256) (&sH[0][0])[i] = 0.f;
  float cfr[2][8];
#pragma unroll
  for (int q = 0; q < 2; ++q)
#pragma unroll
    for (int v = 0; v < 8; ++v) cfr[q][v] = 0.f;
  __syncthreads();

  for (int t = 0; t < T_STEPS; ++t) {
    float num1 = 0.f, num2 = 0.f, num3 = 0.f, den = 0.f;

    // phase 0: load x/m/d tile, gamma_x, mask sum
    for (int i = tid; i < 16 * FD; i += 256) {
      int r = i >> 7, c = i & 127;
      size_t g = ((size_t)(b0 + r) * T_STEPS + t) * FD + c;
      float xv = X[g], mv = Mm[g], dv = Dd[g];
      sX[r][c] = xv;
      sM[r][c] = mv;
      sD[r][c] = (__bf16)dv;
      sGX[r][c] = __expf(-fmaxf(dv * wdx[c] + bdx[c], 0.f));
      den += mv;
    }
    __syncthreads();

    // phase 1: gamma_h = exp(-relu(d @ Wdh^T + bdh)); h *= gamma_h
    for (int nt = wave; nt < 16; nt += 8) {
      v8f acc = {};
#pragma unroll
      for (int kt = 0; kt < 4; ++kt) {
        v16bf a = a_frag_bf(&sD[0][0], FD, kt * 32, lane);
        v16bf b = b_frag(Wdh_b, FD, nt * 16, kt * 32, lane);
        acc = wmma_bf16(a, b, acc);
      }
      int col = nt * 16 + nlo;
      float bv = bdh[col];
#pragma unroll
      for (int v = 0; v < 8; ++v)
        sH[mb + v][col] *= __expf(-fmaxf(acc[v] + bv, 0.f));
    }
    __syncthreads();

    // phase 2: x_h = h @ Whist^T + bhist ; loss term 1
    {
      int nt = wave;
      v8f acc = {};
#pragma unroll
      for (int kt = 0; kt < 8; ++kt) {
        v16bf a = a_frag_f32(&sH[0][0], HD, kt * 32, lane);
        v16bf b = b_frag(Whist_b, HD, nt * 16, kt * 32, lane);
        acc = wmma_bf16(a, b, acc);
      }
      int col = nt * 16 + nlo;
      float bv = bhist[col];
#pragma unroll
      for (int v = 0; v < 8; ++v) {
        float xh = acc[v] + bv;
        sXH[mb + v][col] = xh;
        num1 += fabsf(xh - sX[mb + v][col]) * sM[mb + v][col];
      }
    }
    __syncthreads();

    // phase 3: z_h = (m*x+(1-m)*x_h) @ Wfeat_m^T + bfeat ; loss term 2
    {
      int nt = wave;
      v8f acc = {};
#pragma unroll
      for (int kt = 0; kt < 4; ++kt) {
        v16bf a = a_frag_mix(&sM[0][0], &sX[0][0], &sXH[0][0], FD, kt * 32, lane);
        v16bf b = b_frag(Wfeat_b, FD, nt * 16, kt * 32, lane);
        acc = wmma_bf16(a, b, acc);
      }
      int col = nt * 16 + nlo;
      float bv = bfeat[col];
#pragma unroll
      for (int v = 0; v < 8; ++v) {
        float zh = acc[v] + bv;
        sZH[mb + v][col] = zh;
        num2 += fabsf(zh - sX[mb + v][col]) * sM[mb + v][col];
      }
    }
    __syncthreads();

    // phase 4: alpha = sigmoid([gamma_x, m] @ Wcomb^T + bcomb); c_h blend; loss 3
    {
      int nt = wave;
      v8f acc = {};
#pragma unroll
      for (int kt = 0; kt < 8; ++kt) {
        v16bf a = (kt < 4) ? a_frag_f32(&sGX[0][0], FD, kt * 32, lane)
                           : a_frag_f32(&sM[0][0], FD, kt * 32 - FD, lane);
        v16bf b = b_frag(Wcomb_b, 2 * FD, nt * 16, kt * 32, lane);
        acc = wmma_bf16(a, b, acc);
      }
      int col = nt * 16 + nlo;
      float bv = bcomb[col];
#pragma unroll
      for (int v = 0; v < 8; ++v) {
        int m_ = mb + v;
        float al = sigm(acc[v] + bv);
        float ch = al * sZH[m_][col] + (1.f - al) * sXH[m_][col];
        sXH[m_][col] = ch;  // in-place: this lane owns (m_,col)
        num3 += fabsf(ch - sX[m_][col]) * sM[m_][col];
      }
    }
    __syncthreads();

    // phase 5: LSTM gates = [c_c,m] @ Wih^T + h @ Whh^T + (bih+bhh)
    float hn[2][8];
    for (int q = 0; q < 2; ++q) {
      int jt = wave + 8 * q;
      v8f ai = {}, af = {}, ag = {}, ao = {};
#pragma unroll
      for (int kt = 0; kt < 8; ++kt) {
        v16bf a = (kt < 4)
                      ? a_frag_mix(&sM[0][0], &sX[0][0], &sXH[0][0], FD, kt * 32, lane)
                      : a_frag_f32(&sM[0][0], FD, kt * 32 - FD, lane);
        ai = wmma_bf16(a, b_frag(Wih_b, 2 * FD, 0 * HD + jt * 16, kt * 32, lane), ai);
        af = wmma_bf16(a, b_frag(Wih_b, 2 * FD, 1 * HD + jt * 16, kt * 32, lane), af);
        ag = wmma_bf16(a, b_frag(Wih_b, 2 * FD, 2 * HD + jt * 16, kt * 32, lane), ag);
        ao = wmma_bf16(a, b_frag(Wih_b, 2 * FD, 3 * HD + jt * 16, kt * 32, lane), ao);
        v16bf ah = a_frag_f32(&sH[0][0], HD, kt * 32, lane);
        ai = wmma_bf16(ah, b_frag(Whh_b, HD, 0 * HD + jt * 16, kt * 32, lane), ai);
        af = wmma_bf16(ah, b_frag(Whh_b, HD, 1 * HD + jt * 16, kt * 32, lane), af);
        ag = wmma_bf16(ah, b_frag(Whh_b, HD, 2 * HD + jt * 16, kt * 32, lane), ag);
        ao = wmma_bf16(ah, b_frag(Whh_b, HD, 3 * HD + jt * 16, kt * 32, lane), ao);
      }
      int col = jt * 16 + nlo;
      float bi = bsum[col], bf_ = bsum[HD + col];
      float bg = bsum[2 * HD + col], bo = bsum[3 * HD + col];
#pragma unroll
      for (int v = 0; v < 8; ++v) {
        float ig = sigm(ai[v] + bi);
        float fg = sigm(af[v] + bf_);
        float gg = tanhf(ag[v] + bg);
        float og = sigm(ao[v] + bo);
        float cn = fg * cfr[q][v] + ig * gg;
        cfr[q][v] = cn;
        hn[q][v] = og * tanhf(cn);
      }
    }
    __syncthreads();  // all reads of sH complete before overwrite

    for (int q = 0; q < 2; ++q) {
      int col = (wave + 8 * q) * 16 + nlo;
#pragma unroll
      for (int v = 0; v < 8; ++v) sH[mb + v][col] = hn[q][v];
    }
    __syncthreads();

    // imputed output (coalesced), final h, loss reduction
    for (int i = tid; i < 16 * FD; i += 256) {
      int r = i >> 7, c = i & 127;
      float mv = sM[r][c];
      size_t g = ((size_t)(b0 + r) * T_STEPS + t) * FD + c;
      out[g] = mv * sX[r][c] + (1.f - mv) * sXH[r][c];
    }
    if (t == T_STEPS - 1) {
      for (int i = tid; i < 16 * HD; i += 256) {
        int r = i >> 8, c = i & 255;
        out_h[(size_t)(b0 + r) * HD + c] = sH[r][c];
      }
    }
    float v1 = num1, v2 = num2, v3 = num3, v4 = den;
    for (int off = 16; off > 0; off >>= 1) {
      v1 += __shfl_down(v1, off, 32);
      v2 += __shfl_down(v2, off, 32);
      v3 += __shfl_down(v3, off, 32);
      v4 += __shfl_down(v4, off, 32);
    }
    if (lane == 0) {
      atomicAdd(&acc_g[t * 4 + 0], v1);
      atomicAdd(&acc_g[t * 4 + 1], v2);
      atomicAdd(&acc_g[t * 4 + 2], v3);
      atomicAdd(&acc_g[t * 4 + 3], v4);
    }
    __syncthreads();
  }
}

// ---------------------------------------------------------------------------
__global__ void rits_loss(const float* __restrict__ acc, float* __restrict__ out_loss) {
  if (threadIdx.x == 0 && blockIdx.x == 0) {
    float loss = 0.f;
    for (int t = 0; t < T_STEPS; ++t) {
      float den = acc[t * 4 + 3] + 1e-9f;
      loss += (acc[t * 4 + 0] + acc[t * 4 + 1] + acc[t * 4 + 2]) / den;
    }
    out_loss[0] = loss / (3.0f * T_STEPS);
  }
}

// ---------------------------------------------------------------------------
extern "C" void kernel_launch(void* const* d_in, const int* in_sizes, int n_in,
                              void* d_out, int out_size, void* d_ws, size_t ws_size,
                              hipStream_t stream) {
  const float* X     = (const float*)d_in[0];
  const float* Mm    = (const float*)d_in[1];
  const float* Dd    = (const float*)d_in[2];
  const float* Wdh   = (const float*)d_in[3];
  const float* bdh   = (const float*)d_in[4];
  const float* Wdx   = (const float*)d_in[5];
  const float* bdx   = (const float*)d_in[6];
  const float* Whist = (const float*)d_in[7];
  const float* bhist = (const float*)d_in[8];
  const float* Wfeat = (const float*)d_in[9];
  const float* bfeat = (const float*)d_in[10];
  const float* Wcomb = (const float*)d_in[11];
  const float* bcomb = (const float*)d_in[12];
  const float* Wih   = (const float*)d_in[13];
  const float* Whh   = (const float*)d_in[14];
  const float* bih   = (const float*)d_in[15];
  const float* bhh   = (const float*)d_in[16];
  char* ws = (char*)d_ws;
  float* out = (float*)d_out;

  rits_prep<<<256, 256, 0, stream>>>(Wdh, Whist, Wfeat, Wcomb, Wih, Whh, Wdx, bih, bhh, ws);
  rits_main<<<NB / 16, 256, 0, stream>>>(X, Mm, Dd, bdh, bdx, bhist, bfeat, bcomb, ws, out);
  rits_loss<<<1, 32, 0, stream>>>((const float*)ws,
                                  out + (size_t)NB * T_STEPS * FD + (size_t)NB * HD);
}
